// AnchorDeformAtt_53755810677160
// MI455X (gfx1250) — matrix operand order repaired and checked by reading, not compiled, not run on hardware
//
#include <hip/hip_runtime.h>
#include <hip/hip_bf16.h>

// ---------------------------------------------------------------------------
// Problem constants (fixed by setup_inputs)
// ---------------------------------------------------------------------------
#define BATCH 2
#define CCH   256
#define HH    64
#define WW    96
#define LL    (HH * WW)          // 6144
#define NHD   8                  // heads
#define NPT   16                 // points
#define HD    32                 // head dim
#define RTOT  656                // 256(value)+16(size)+256(anchor)+128(att)
#define ROW_SIZE   256           // offset of size rows in Yall
#define ROW_ANCH   272
#define ROW_ATT    528
#define NTILE (LL / 16)          // 384 column tiles of 16
#define CTG   (LL / 64 / 8)      // 12 column groups of 8 waves * 64 cols
#define KKN   (CCH / 32)         // 8 K-steps

typedef __attribute__((ext_vector_type(16))) __bf16         v16bf;
typedef __attribute__((ext_vector_type(16))) unsigned short v16u;
typedef __attribute__((ext_vector_type(8)))  unsigned short ushort8;
typedef __attribute__((ext_vector_type(8)))  float          v8f;
typedef __attribute__((ext_vector_type(4)))  unsigned int   uint32x4;
typedef __attribute__((ext_vector_type(8)))  int            int32x8;
typedef __attribute__((ext_vector_type(4)))  int            int32x4;

__device__ __forceinline__ unsigned short f2bf(float f) {
    unsigned int u = __builtin_bit_cast(unsigned int, f);
    unsigned int r = u + 0x7FFFu + ((u >> 16) & 1u);   // round-to-nearest-even
    return (unsigned short)(r >> 16);
}

__device__ __forceinline__ float sigmf(float x) {
    return 1.0f / (1.0f + __expf(-x));
}

// ---------------------------------------------------------------------------
// Kernel 1: pack all conv1x1 weights into one 656x256 bf16 matrix (+bias),
// and convert w_out to bf16.
// ---------------------------------------------------------------------------
__global__ __launch_bounds__(256)
void pack_weights(const float* __restrict__ w_value, const float* __restrict__ w_size,
                  const float* __restrict__ w_anchor, const float* __restrict__ w_att,
                  const float* __restrict__ b_value, const float* __restrict__ b_size,
                  const float* __restrict__ b_anchor, const float* __restrict__ b_att,
                  const float* __restrict__ w_out,
                  unsigned short* __restrict__ Wcat, float* __restrict__ bcat,
                  unsigned short* __restrict__ WoutBf) {
    int tid = blockIdx.x * blockDim.x + threadIdx.x;
    const int N1 = RTOT * CCH;
    if (tid < N1) {
        int r = tid >> 8, c = tid & 255;
        float v;
        if      (r < ROW_SIZE) v = w_value [(r            ) * CCH + c];
        else if (r < ROW_ANCH) v = w_size  [(r - ROW_SIZE ) * CCH + c];
        else if (r < ROW_ATT ) v = w_anchor[(r - ROW_ANCH ) * CCH + c];
        else                   v = w_att   [(r - ROW_ATT  ) * CCH + c];
        Wcat[tid] = f2bf(v);
    }
    if (tid < CCH * CCH) WoutBf[tid] = f2bf(w_out[tid]);
    if (tid < RTOT) {
        float bv;
        if      (tid < ROW_SIZE) bv = b_value [tid];
        else if (tid < ROW_ANCH) bv = b_size  [tid - ROW_SIZE];
        else if (tid < ROW_ATT ) bv = b_anchor[tid - ROW_ANCH];
        else                     bv = b_att   [tid - ROW_ATT];
        bcat[tid] = bv;
    }
}

// ---------------------------------------------------------------------------
// Kernel 2: feature map f32 -> bf16, stored in WMMA-B swizzled layout:
// per batch, per 32-row K-block kk, per 16-col tile nblk: 512 elems laid out
// [n(0..15)][kin(0..31)] contiguous.  Each B-fragment lane read becomes two
// contiguous 16B loads; a wave streams each tile as one contiguous 1KB region.
// ---------------------------------------------------------------------------
__device__ __forceinline__ size_t swz_idx(int b, int c, int l) {
    int kk = c >> 5, kin = c & 31, nblk = l >> 4, n = l & 15;
    return (((size_t)(b * (CCH / 32) + kk) * NTILE + nblk) << 9) + (n << 5) + kin;
}

__global__ __launch_bounds__(256)
void swizzle_feat(const float* __restrict__ feat, unsigned short* __restrict__ Xswz, int n) {
    int tid = blockIdx.x * blockDim.x + threadIdx.x;
    if (tid >= n) return;
    int l = tid % LL; int t = tid / LL;
    int c = t % CCH;  int b = t / CCH;
    Xswz[swz_idx(b, c, l)] = f2bf(feat[tid]);
}

// ---------------------------------------------------------------------------
// Stage a 16x256 bf16 weight tile (8KB, contiguous in global) into LDS.
// Preferred path: Tensor Data Mover (TENSOR_LOAD_TO_LDS) issued by wave 0,
// completion via s_wait_tensorcnt, then workgroup barrier.
// D# per CDNA5 ISA ch.8: group0 {count=1, lds_addr, global_addr, type=2},
// group1 {data_size=4B, tensor_dim0=tile_dim0=2048, tensor_dim1=tile_dim1=1}.
// This toolchain exposes the 6-arg builtin:
//   (uint32x4 g0, int32x8 g1, int32x4 g2, int32x4 g3, int32x8 g4, i32 cpol)
// ---------------------------------------------------------------------------
#if __has_builtin(__builtin_amdgcn_tensor_load_to_lds) && __has_builtin(__builtin_amdgcn_s_wait_tensorcnt)
#define USE_TDM 1
#endif

__device__ __forceinline__
void stage_Atile(const unsigned short* __restrict__ Wsrc, int rowbase,
                 unsigned short* __restrict__ lds) {
#ifdef USE_TDM
    if ((threadIdx.x >> 5) == 0) {
        unsigned long long gaddr = (unsigned long long)(Wsrc + (size_t)rowbase * CCH);
        unsigned int lds_off = (unsigned int)(unsigned long long)lds;  // low 32 bits = LDS byte addr
        uint32x4 g0;
        g0[0] = 1u;                                        // count=1 (valid descriptor)
        g0[1] = lds_off;                                   // lds_addr [63:32]
        g0[2] = (unsigned int)gaddr;                       // global_addr [95:64]
        g0[3] = ((unsigned int)(gaddr >> 32) & 0x01FFFFFFu) | 0x80000000u; // addr[56:32] | type=2
        int32x8 g1;
        g1[0] = (int)0x00020000;   // workgroup_mask=0, data_size=2 (4B)
        g1[1] = (int)0x08000000;   // tensor_dim0[15:0]=2048 at bits 63:48
        g1[2] = (int)0x00010000;   // tensor_dim0 hi=0, tensor_dim1[15:0]=1
        g1[3] = (int)0x08000000;   // tile_dim0=2048 at bits 127:112
        g1[4] = 1;                 // tile_dim1=1, tile_dim2=0
        g1[5] = 2048;              // tensor_dim0_stride low 32
        g1[6] = 0;
        g1[7] = 0;
        int32x4 gz4 = {0, 0, 0, 0};
        int32x8 gz8 = {0, 0, 0, 0, 0, 0, 0, 0};
        __builtin_amdgcn_tensor_load_to_lds(g0, g1, gz4, gz4, gz8, 0);
        __builtin_amdgcn_s_wait_tensorcnt(0);
    }
    __syncthreads();
#else
    const uint4* src = (const uint4*)(Wsrc + (size_t)rowbase * CCH);
    uint4* dst = (uint4*)lds;
    for (int i = threadIdx.x; i < 512; i += blockDim.x) dst[i] = src[i];  // 8KB
    __syncthreads();
#endif
}

// ---------------------------------------------------------------------------
// Fragment loads for one K-step (32 K-values): A from LDS (2 x ds_load_b128),
// B for 4 N-subtiles from swizzled global (8 x global_load_b128).
// ---------------------------------------------------------------------------
__device__ __forceinline__
void load_frags(const unsigned short* __restrict__ arow,   // LDS row base (lane's M row)
                const unsigned short* __restrict__ bb,     // lane's B base (kk=0)
                int kk, int kloA, v16u& au, v16u bu[4]) {
    const int k = kk * 32;
    ushort8 a0 = *(const ushort8*)(arow + k + kloA);
    ushort8 a1 = *(const ushort8*)(arow + k + kloA + 8);
#pragma unroll
    for (int e = 0; e < 8; ++e) { au[e] = a0[e]; au[e + 8] = a1[e]; }
    const unsigned short* bbase = bb + (((size_t)kk * NTILE) << 9);
#pragma unroll
    for (int s = 0; s < 4; ++s) {
        const unsigned short* bp = bbase + ((size_t)s << 9);
        ushort8 b0 = *(const ushort8*)bp;
        ushort8 b1 = *(const ushort8*)(bp + 8);
#pragma unroll
        for (int e = 0; e < 8; ++e) { bu[s][e] = b0[e]; bu[s][e + 8] = b1[e]; }
    }
}

// ---------------------------------------------------------------------------
// WMMA core: one wave computes a 16(M) x 64(N) f32 tile, software-pipelined:
// fragments for K-step kk+1 are loaded before the WMMAs of step kk issue, so
// global-load latency overlaps matrix math instead of stalling at s_wait.
// ---------------------------------------------------------------------------
__device__ __forceinline__
void mma_tile16x64(const unsigned short* __restrict__ Atile,  // 16x256 bf16 in LDS
                   const unsigned short* __restrict__ Xswz,   // per-batch swizzled B
                   int colbase, v8f acc[4]) {
    const int lane = threadIdx.x & 31;
    const int m    = lane & 15;
    const int hi   = (lane >> 4) & 1;
    const int kloA = hi * 16;
    const int kb   = hi * 16;
    const int n    = lane & 15;
    const unsigned short* arow = Atile + m * CCH;
    const unsigned short* bb   = Xswz + (((size_t)(colbase >> 4)) << 9) + (n << 5) + kb;

    v16u au2[2];
    v16u bu2[2][4];
    load_frags(arow, bb, 0, kloA, au2[0], bu2[0]);

#pragma unroll
    for (int kk = 0; kk < KKN; ++kk) {
        const int cur = kk & 1, nxt = cur ^ 1;
        if (kk + 1 < KKN)
            load_frags(arow, bb, kk + 1, kloA, au2[nxt], bu2[nxt]);
        if (kk + 2 < KKN)   // pull the slab after next toward near cache
            __builtin_prefetch(bb + (((size_t)(kk + 2) * NTILE) << 9), 0, 3);
        v16bf af = __builtin_bit_cast(v16bf, au2[cur]);
#pragma unroll
        for (int s = 0; s < 4; ++s) {
            acc[s] = __builtin_amdgcn_wmma_f32_16x16x32_bf16(false, af, false,
                     __builtin_bit_cast(v16bf, bu2[cur][s]), (short)0, acc[s], false, false);
        }
    }
}

// ---------------------------------------------------------------------------
// Kernel 3: fused projection GEMM -> Yall (B, 656, L) f32 with bias.
// Block = (b, rt, column-group of 8 waves x 64 cols). 2*41*12 = 984 blocks.
// All 8 waves share one TDM-staged 16x256 weight tile in LDS.
// ---------------------------------------------------------------------------
__global__ __launch_bounds__(256)
void gemm_proj(const unsigned short* __restrict__ Wcat,
               const unsigned short* __restrict__ Xswz,
               const float* __restrict__ bcat,
               float* __restrict__ Yall) {
    __shared__ unsigned short Atile[16 * CCH];
    int bid = blockIdx.x;
    int ctg = bid % CTG;  bid /= CTG;
    int rt  = bid % (RTOT / 16);
    int b   = bid / (RTOT / 16);
    if (b >= BATCH) return;

    stage_Atile(Wcat, rt * 16, Atile);

    int ct = ctg * 8 + (threadIdx.x >> 5);
    v8f acc[4] = {};
    mma_tile16x64(Atile, Xswz + (size_t)b * CCH * LL, ct * 64, acc);

    const int lane = threadIdx.x & 31;
    const int n    = lane & 15;
    const int mrow = ((lane >> 4) & 1) * 8;
#pragma unroll
    for (int s = 0; s < 4; ++s) {
#pragma unroll
        for (int r = 0; r < 8; ++r) {
            int row = rt * 16 + mrow + r;
            int col = ct * 64 + s * 16 + n;
            Yall[((size_t)b * RTOT + row) * LL + col] = acc[s][r] + bcat[row];
        }
    }
}

// ---------------------------------------------------------------------------
// Kernel 4: value tensor (B,C,L) -> (B,NH,L,HD) for contiguous bilinear taps
// ---------------------------------------------------------------------------
__global__ __launch_bounds__(256)
void transpose_mem(const float* __restrict__ Yall, float* __restrict__ memT, int n) {
    int tid = blockIdx.x * blockDim.x + threadIdx.x;
    if (tid >= n) return;
    int d  = tid & (HD - 1);
    int t  = tid >> 5;
    int l  = t % LL;  t /= LL;
    int h  = t & (NHD - 1);
    int b  = t >> 3;
    memT[tid] = Yall[((size_t)b * RTOT + h * HD + d) * LL + l];
}

// ---------------------------------------------------------------------------
// Kernel 5: anchor-deformable sampling. One thread per (b, h, l):
// sigmoid/clip window size, sigmoid offsets, softmax over 16 points,
// 4-tap bilinear over 32 contiguous channels. Output written bf16 directly in
// the WMMA-B swizzled layout (c-block == head), 64 contiguous bytes/thread.
// ---------------------------------------------------------------------------
__global__ __launch_bounds__(256)
void sample_att(const float* __restrict__ Yall, const float* __restrict__ memT,
                unsigned short* __restrict__ Samp) {
    const int L = LL;
    int tid = blockIdx.x * blockDim.x + threadIdx.x;
    int l = tid % L;  int t = tid / L;
    int h = t & (NHD - 1);
    int b = t >> 3;
    if (b >= BATCH) return;

    const float* Yb = Yall + (size_t)b * RTOT * L + l;   // row r at Yb[r*L]
    int iy = l / WW, ix = l - iy * WW;
    float cx = (ix + 0.5f) / (WW + 1e-6f);
    float cy = (iy + 0.5f) / (HH + 1e-6f);

    float sx = fminf(fmaxf(sigmf(Yb[(size_t)(ROW_SIZE + h * 2 + 0) * L]), 0.25f), 0.75f);
    float sy = fminf(fmaxf(sigmf(Yb[(size_t)(ROW_SIZE + h * 2 + 1) * L]), 0.25f), 0.75f);

    // softmax over the 16 attention logits
    float lg[NPT];
    float mx = -3.4e38f;
#pragma unroll
    for (int p = 0; p < NPT; ++p) {
        lg[p] = Yb[(size_t)(ROW_ATT + h * NPT + p) * L];
        mx = fmaxf(mx, lg[p]);
    }
    float se = 0.f;
#pragma unroll
    for (int p = 0; p < NPT; ++p) { lg[p] = __expf(lg[p] - mx); se += lg[p]; }
    float inv = 1.0f / se;

    float acc[HD];
#pragma unroll
    for (int d = 0; d < HD; ++d) acc[d] = 0.f;

    const float* vb = memT + (size_t)(b * NHD + h) * L * HD;

    for (int p = 0; p < NPT; ++p) {
        float ox = sigmf(Yb[(size_t)(ROW_ANCH + (h * NPT + p) * 2 + 0) * L]);
        float oy = sigmf(Yb[(size_t)(ROW_ANCH + (h * NPT + p) * 2 + 1) * L]);
        float gx = fminf(fmaxf(cx - 0.5f * sx + ox * sx, 0.f), 1.f) * (float)(WW - 1);
        float gy = fminf(fmaxf(cy - 0.5f * sy + oy * sy, 0.f), 1.f) * (float)(HH - 1);
        float fx = floorf(gx), fy = floorf(gy);
        float wx = gx - fx,    wy = gy - fy;
        int x0 = (int)fx,      y0 = (int)fy;
        int x0i = min(x0, WW - 1), x1i = min(x0 + 1, WW - 1);
        int y0i = min(y0, HH - 1), y1i = min(y0 + 1, HH - 1);

        float a   = lg[p] * inv;
        float w00 = a * (1.f - wy) * (1.f - wx);
        float w01 = a * (1.f - wy) * wx;
        float w10 = a * wy * (1.f - wx);
        float w11 = a * wy * wx;

        const float4* p00 = (const float4*)(vb + (size_t)(y0i * WW + x0i) * HD);
        const float4* p01 = (const float4*)(vb + (size_t)(y0i * WW + x1i) * HD);
        const float4* p10 = (const float4*)(vb + (size_t)(y1i * WW + x0i) * HD);
        const float4* p11 = (const float4*)(vb + (size_t)(y1i * WW + x1i) * HD);
#pragma unroll
        for (int dd = 0; dd < HD / 4; ++dd) {
            float4 v00 = p00[dd], v01 = p01[dd], v10 = p10[dd], v11 = p11[dd];
            acc[4 * dd + 0] += w00 * v00.x + w01 * v01.x + w10 * v10.x + w11 * v11.x;
            acc[4 * dd + 1] += w00 * v00.y + w01 * v01.y + w10 * v10.y + w11 * v11.y;
            acc[4 * dd + 2] += w00 * v00.z + w01 * v01.z + w10 * v10.z + w11 * v11.z;
            acc[4 * dd + 3] += w00 * v00.w + w01 * v01.w + w10 * v10.w + w11 * v11.w;
        }
    }

    // write 32 bf16 = 64 contiguous bytes in swizzled-B layout (kk == h)
    ushort8 ov[4];
#pragma unroll
    for (int d = 0; d < HD; ++d) ov[d >> 3][d & 7] = f2bf(acc[d]);
    ushort8* dst = (ushort8*)(Samp + swz_idx(b, h * HD, l));
#pragma unroll
    for (int q = 0; q < 4; ++q) dst[q] = ov[q];
}

// ---------------------------------------------------------------------------
// Kernel 6: output GEMM (w_out x sampled) with fused BatchNorm -> d_out f32
// Blocks = 2*16*12 = 384, 8 waves each; LDS-staged weight tile as in gemm_proj.
// ---------------------------------------------------------------------------
__global__ __launch_bounds__(256)
void gemm_out_bn(const unsigned short* __restrict__ WoutBf,
                 const unsigned short* __restrict__ Samp,
                 const float* __restrict__ gamma, const float* __restrict__ beta,
                 const float* __restrict__ mean,  const float* __restrict__ var,
                 float* __restrict__ Out) {
    __shared__ unsigned short Atile[16 * CCH];
    int bid = blockIdx.x;
    int ctg = bid % CTG;  bid /= CTG;
    int rt  = bid % (CCH / 16);
    int b   = bid / (CCH / 16);
    if (b >= BATCH) return;

    stage_Atile(WoutBf, rt * 16, Atile);

    int ct = ctg * 8 + (threadIdx.x >> 5);
    v8f acc[4] = {};
    mma_tile16x64(Atile, Samp + (size_t)b * CCH * LL, ct * 64, acc);

    const int lane = threadIdx.x & 31;
    const int n    = lane & 15;
    const int mrow = ((lane >> 4) & 1) * 8;
#pragma unroll
    for (int s = 0; s < 4; ++s) {
#pragma unroll
        for (int r = 0; r < 8; ++r) {
            int o   = rt * 16 + mrow + r;
            int col = ct * 64 + s * 16 + n;
            float sc = gamma[o] * __frsqrt_rn(var[o] + 1e-5f);
            Out[((size_t)b * CCH + o) * LL + col] = (acc[s][r] - mean[o]) * sc + beta[o];
        }
    }
}

// ---------------------------------------------------------------------------
// Launcher
// ---------------------------------------------------------------------------
extern "C" void kernel_launch(void* const* d_in, const int* in_sizes, int n_in,
                              void* d_out, int out_size, void* d_ws, size_t ws_size,
                              hipStream_t stream) {
    const float* feat     = (const float*)d_in[0];
    const float* w_size   = (const float*)d_in[1];
    const float* b_size   = (const float*)d_in[2];
    const float* w_anchor = (const float*)d_in[3];
    const float* b_anchor = (const float*)d_in[4];
    const float* w_value  = (const float*)d_in[5];
    const float* b_value  = (const float*)d_in[6];
    const float* w_att    = (const float*)d_in[7];
    const float* b_att    = (const float*)d_in[8];
    const float* w_out    = (const float*)d_in[9];
    const float* bn_gamma = (const float*)d_in[10];
    const float* bn_beta  = (const float*)d_in[11];
    const float* bn_mean  = (const float*)d_in[12];
    const float* bn_var   = (const float*)d_in[13];
    float* out = (float*)d_out;

    // workspace carve-up (256B aligned)
    char*  ws  = (char*)d_ws;
    size_t off = 0;
    auto carve = [&](size_t bytes) { char* p = ws + off; off += (bytes + 255) & ~(size_t)255; return p; };
    unsigned short* Xswz   = (unsigned short*)carve((size_t)BATCH * CCH * LL * 2);   // 6.3 MB
    unsigned short* Wcat   = (unsigned short*)carve((size_t)RTOT * CCH * 2);         // 0.34 MB
    float*          bcat   = (float*)         carve((size_t)RTOT * 4);
    float*          Yall   = (float*)         carve((size_t)BATCH * RTOT * LL * 4);  // 32.3 MB
    float*          memT   = (float*)         carve((size_t)BATCH * NHD * LL * HD * 4); // 12.6 MB
    unsigned short* Samp   = (unsigned short*)carve((size_t)BATCH * CCH * LL * 2);   // 6.3 MB
    unsigned short* WoutBf = (unsigned short*)carve((size_t)CCH * CCH * 2);          // 0.13 MB
    (void)ws_size; (void)in_sizes; (void)n_in; (void)out_size;

    // 1. pack weights / biases
    {
        int n = RTOT * CCH;
        pack_weights<<<(n + 255) / 256, 256, 0, stream>>>(
            w_value, w_size, w_anchor, w_att, b_value, b_size, b_anchor, b_att,
            w_out, Wcat, bcat, WoutBf);
    }
    // 2. feature f32 -> bf16 swizzled B layout
    {
        int n = BATCH * CCH * LL;
        swizzle_feat<<<(n + 255) / 256, 256, 0, stream>>>(feat, Xswz, n);
    }
    // 3. fused projection GEMM (656 x 256 x 6144 per batch)
    {
        int nblocks = BATCH * (RTOT / 16) * CTG;   // 984
        gemm_proj<<<nblocks, 256, 0, stream>>>(Wcat, Xswz, bcat, Yall);
    }
    // 4. value layout transform
    {
        int n = BATCH * NHD * LL * HD;
        transpose_mem<<<(n + 255) / 256, 256, 0, stream>>>(Yall, memT, n);
    }
    // 5. deformable sampling
    {
        int n = BATCH * NHD * LL;
        sample_att<<<(n + 255) / 256, 256, 0, stream>>>(Yall, memT, Samp);
    }
    // 6. output GEMM + BN
    {
        int nblocks = BATCH * (CCH / 16) * CTG;    // 384
        gemm_out_bn<<<nblocks, 256, 0, stream>>>(
            WoutBf, Samp, bn_gamma, bn_beta, bn_mean, bn_var, out);
    }
}
